// PMClassifier_89601607729186
// MI455X (gfx1250) — matrix operand order repaired and checked by reading, compiled, not run on hardware
//
#include <hip/hip_runtime.h>

// ---------------------------------------------------------------------------
// PMClassifier on MI455X (gfx1250): encoder MLP -> PMFlow ODE -> head MLP.
// The 32x32 encoder layer (~75% of FLOPs) runs on v_wmma_f32_16x16x32_f16.
// Each wave owns NT=8 contiguous 16-row tiles so the WMMA B operands and
// layer-1 weights are built once per wave, not once per tile.
// ---------------------------------------------------------------------------

typedef __attribute__((ext_vector_type(16))) _Float16 v16h;
typedef __attribute__((ext_vector_type(8)))  float    v8f;

#define PM_STEPS 6
#define PM_DT    0.2f
#define PM_EPS   1e-4f
#define NT       8            // 16-row tiles per wave

__device__ __forceinline__ float fast_tanh(float x) {
#if __has_builtin(__builtin_amdgcn_tanhf)
    return __builtin_amdgcn_tanhf(x);              // gfx1250 v_tanh_f32
#else
    float xc = fminf(fmaxf(x, -15.0f), 15.0f);
    float t  = __builtin_amdgcn_exp2f(xc * 2.8853900817779268f); // 2*log2(e)
    return (t - 1.0f) * __builtin_amdgcn_rcpf(t + 1.0f);
#endif
}

__device__ __forceinline__ float rsq_fast(float x) {
#if __has_builtin(__builtin_amdgcn_rsqf)
    return __builtin_amdgcn_rsqf(x);
#else
    return rsqrtf(x);
#endif
}

__device__ __forceinline__ float rcp_fast(float x) {
#if __has_builtin(__builtin_amdgcn_rcpf)
    return __builtin_amdgcn_rcpf(x);
#else
    return 1.0f / x;
#endif
}

// 8 waves / block; each wave processes NT tiles of 16 rows
__global__ __launch_bounds__(256) void pmclassifier_kernel(
    const float* __restrict__ x,
    const float* __restrict__ ew1, const float* __restrict__ eb1,
    const float* __restrict__ ew2, const float* __restrict__ eb2,
    const float* __restrict__ ew3, const float* __restrict__ eb3,
    const float* __restrict__ hw1, const float* __restrict__ hb1,
    const float* __restrict__ hw2, const float* __restrict__ hb2,
    float* __restrict__ out)
{
    // per-wave 16x32 activation tile, row stride padded to 33 floats:
    // 33 coprime with 64 banks -> conflict-free scatter & gather.
    __shared__ float lds[8][16 * 33];

    const int lane = threadIdx.x & 31;
    const int wave = threadIdx.x >> 5;
    const int half = lane >> 4;          // 0: lanes 0-15, 1: lanes 16-31
    const int l16  = lane & 15;
    const int kb   = half * 16;          // this lane-half's K base (layer 3 / head)

    // ================= per-wave invariants (amortized over NT tiles) ========

    // B operands for the layer-2 WMMAs (enc_w2 columns).
    // B 32x16 f16 layout: lane holds column N=lane%16;
    // element e <-> K = (e<8 ? e : e+8) + 8*half (matches A).
    v16h b_lo, b_hi;
    #pragma unroll
    for (int e = 0; e < 16; ++e) {
        const int kk = ((e < 8) ? e : e + 8) + 8 * half;
        b_lo[e] = (_Float16)ew2[kk * 32 + l16];
        b_hi[e] = (_Float16)ew2[kk * 32 + l16 + 16];
    }
    const float bias_lo = eb2[l16];
    const float bias_hi = eb2[l16 + 16];

    // Layer-1 weights in WMMA-A element order:
    // element e <-> hidden dim j = (e<8 ? e : e+8) + 8*half.
    float w1a[16], w1b[16], b1r[16];
    #pragma unroll
    for (int e = 0; e < 16; ++e) {
        const int j = ((e < 8) ? e : e + 8) + 8 * half;
        w1a[e] = ew1[j];
        w1b[e] = ew1[32 + j];
        b1r[e] = eb1[j];
    }

    const int tile0 = (blockIdx.x * 8 + wave) * NT;

    // ================= tile loop ===========================================
    for (int it = 0; it < NT; ++it) {
        const int m0 = (tile0 + it) * 16;
        const int m  = m0 + l16;         // lanes L and L+16 share row m

        // prefetch next tile's inputs while this tile computes
        if (it + 1 < NT)
            __builtin_prefetch(x + 2 * (m + 16), 0, 3);

        // ---- layer 1 (2->32, tanh) straight into WMMA A layout ----
        const float2 xv = *reinterpret_cast<const float2*>(x + 2 * m);
        v16h a;
        #pragma unroll
        for (int e = 0; e < 16; ++e)
            a[e] = (_Float16)fast_tanh(fmaf(xv.x, w1a[e], fmaf(xv.y, w1b[e], b1r[e])));

        // ---- layer 2 (32->32): two V_WMMA_F32_16X16X32_F16, bias in C ----
        v8f c_lo, c_hi;
        #pragma unroll
        for (int r = 0; r < 8; ++r) { c_lo[r] = bias_lo; c_hi[r] = bias_hi; }
        c_lo = __builtin_amdgcn_wmma_f32_16x16x32_f16(false, a, false, b_lo,
                                                      (short)0, c_lo, false, false);
        c_hi = __builtin_amdgcn_wmma_f32_16x16x32_f16(false, a, false, b_hi,
                                                      (short)0, c_hi, false, false);

        // ---- tanh + transpose to [row][col] via LDS ----
        // D layout: VGPR r, lane L -> [M = r + 8*half, N = L%16 (+16 for hi)].
        #pragma unroll
        for (int r = 0; r < 8; ++r) {
            const int row = r + 8 * half;
            lds[wave][row * 33 + l16]      = fast_tanh(c_lo[r]);
            lds[wave][row * 33 + l16 + 16] = fast_tanh(c_hi[r]);
        }
        __syncthreads();

        // ---- layer 3 (32->2): lane pair (L, L+16) splits the K=32 dot ----
        const int row = l16;
        float z0 = 0.0f, z1 = 0.0f;
        #pragma unroll
        for (int t = 0; t < 16; ++t) {
            const int   j = kb + t;
            const float h = lds[wave][row * 33 + j];
            z0 = fmaf(h, ew3[2 * j + 0], z0);
            z1 = fmaf(h, ew3[2 * j + 1], z1);
        }
        z0 += __shfl_xor(z0, 16, 32);  z0 += eb3[0];
        z1 += __shfl_xor(z1, 16, 32);  z1 += eb3[1];

        // ---- PMFlow: 6 explicit-Euler steps of grad ln n(z), clip [-3,3] ----
        #pragma unroll
        for (int s = 0; s < PM_STEPS; ++s) {
            float nacc = 1.0f, gx = 0.0f, gy = 0.0f;
            #pragma unroll
            for (int cc = 0; cc < 2; ++cc) {
                const float cx  = (cc == 0) ? -0.6f : 0.6f;
                const float dx  = z0 - cx;
                const float dy  = z1;
                const float d2  = fmaf(dx, dx, fmaf(dy, dy, PM_EPS));
                const float ir  = rsq_fast(d2);       // 1/r
                const float mir = 0.35f * ir;         // mu/r
                nacc += mir;
                const float s3 = mir * ir * ir;       // mu/r^3
                gx = fmaf(-s3, dx, gx);
                gy = fmaf(-s3, dy, gy);
            }
            const float dtn = PM_DT * rcp_fast(nacc);
            z0 = fminf(fmaxf(fmaf(dtn, gx, z0), -3.0f), 3.0f);
            z1 = fminf(fmaxf(fmaf(dtn, gy, z1), -3.0f), 3.0f);
        }

        // ---- head MLP (2->32 tanh, 32->2), split across the lane pair ----
        float p0 = 0.0f, p1 = 0.0f;
        #pragma unroll
        for (int t = 0; t < 16; ++t) {
            const int   j = kb + t;
            const float h = fast_tanh(fmaf(z0, hw1[j], fmaf(z1, hw1[32 + j], hb1[j])));
            p0 = fmaf(h, hw2[2 * j + 0], p0);
            p1 = fmaf(h, hw2[2 * j + 1], p1);
        }
        p0 += __shfl_xor(p0, 16, 32);
        p1 += __shfl_xor(p1, 16, 32);

        if (half == 0) {
            float2 res = make_float2(p0 + hb2[0], p1 + hb2[1]);
            *reinterpret_cast<float2*>(out + 2 * (m0 + row)) = res;
        }

        // keep next iteration's LDS stores behind this iteration's loads
        __syncthreads();
    }
}

extern "C" void kernel_launch(void* const* d_in, const int* in_sizes, int n_in,
                              void* d_out, int out_size, void* d_ws, size_t ws_size,
                              hipStream_t stream) {
    (void)n_in; (void)out_size; (void)d_ws; (void)ws_size;
    const float* x   = (const float*)d_in[0];
    const float* ew1 = (const float*)d_in[1];
    const float* eb1 = (const float*)d_in[2];
    const float* ew2 = (const float*)d_in[3];
    const float* eb2 = (const float*)d_in[4];
    const float* ew3 = (const float*)d_in[5];
    const float* eb3 = (const float*)d_in[6];
    const float* hw1 = (const float*)d_in[7];
    const float* hb1 = (const float*)d_in[8];
    const float* hw2 = (const float*)d_in[9];
    const float* hb2 = (const float*)d_in[10];
    float* out = (float*)d_out;

    const int B = in_sizes[0] / 2;            // 2,097,152
    const int rows_per_block = 16 * NT * 8;   // 1024 rows / block
    const int blocks = B / rows_per_block;    // 2048

    pmclassifier_kernel<<<blocks, 256, 0, stream>>>(
        x, ew1, eb1, ew2, eb2, ew3, eb3, hw1, hb1, hw2, hb2, out);
}
